// RegionProposalNet_47407849013559
// MI455X (gfx1250) — matrix operand order repaired and checked by reading, not compiled, not run on hardware
//
#include <hip/hip_runtime.h>
#include <math.h>

// ---------------------------------------------------------------------------
// Types for CDNA5 WMMA (wave32, v_wmma_f32_16x16x32_bf16) + TDM descriptors
// ---------------------------------------------------------------------------
typedef __attribute__((ext_vector_type(16))) __bf16 v16bf;
typedef __attribute__((ext_vector_type(8)))  float  v8f;
typedef __attribute__((ext_vector_type(4)))  unsigned int u32x4;
typedef __attribute__((ext_vector_type(8)))  int    i32x8;
typedef __attribute__((ext_vector_type(4)))  int    i32x4;

union Frag {                // 32 bytes: 8 VGPRs per lane
    unsigned int d[8];
    v16bf        b;
};

__device__ inline v8f wmma_bf16(v16bf a, v16bf b, v8f c) {
    return __builtin_amdgcn_wmma_f32_16x16x32_bf16(
        false, a, false, b, (short)0, c, false, false);
}

__device__ inline unsigned short f2bf(float f) {
    unsigned int u = __float_as_uint(f);
    u += 0x7FFFu + ((u >> 16) & 1u);            // round-to-nearest-even
    return (unsigned short)(u >> 16);
}
__device__ inline float bf2f(unsigned short h) {
    return __uint_as_float(((unsigned int)h) << 16);
}

#define INF_F  (__builtin_inff())
#define TOTA   65472          // total anchors per batch (21824 * 3)
#define CLS_BASE 3000         // rois [2,300,5] first
#define REG_BASE 133944       // 3000 + 2*65472

// ---------------------------------------------------------------------------
// TDM: load one 512-row x 32-col bf16 tile of Wp (row stride 2304 elems) into
// LDS with 16B padding after every 64B row (-> 80B LDS row stride).
// D# layout per CDNA5 ISA.
// Toolchain uses the 6-arg builtin: (g0, g1, g2, g3, g4, cpol).
// ---------------------------------------------------------------------------
__device__ inline void tdm_load_B(const unsigned short* gsrc, unsigned lds_off) {
    unsigned long long ga = (unsigned long long)(size_t)gsrc;
    u32x4 g0;
    g0[0] = 1u;                                        // count=1, user mode
    g0[1] = lds_off;                                   // lds_addr (bytes)
    g0[2] = (unsigned)(ga & 0xFFFFFFFFu);              // global_addr[31:0]
    g0[3] = (unsigned)((ga >> 32) & 0x01FFFFFFu)       // global_addr[56:32]
          | (2u << 30);                                // type=2 ("image")
    i32x8 g1;
    g1[0] = (1 << 16)       // data_size = 1 -> 2 bytes
          | (1 << 20)       // pad_enable
          | (3 << 22)       // pad_interval: 16 DWORDs (64B) between pads
          | (3 << 25);      // pad_amount: 4 DWORDs (16B)
    g1[1] = (int)(2304u << 16);        // tensor_dim0[15:0] in bits[31:16]
    g1[2] = (int)(512u << 16);         // tensor_dim0[31:16]=0 | tensor_dim1[15:0]
    g1[3] = (int)(32u << 16);          // tensor_dim1[31:16]=0 | tile_dim0=32
    g1[4] = (int)512;                  // tile_dim1=512 | tile_dim2=0
    g1[5] = (int)2304;                 // tensor_dim0_stride[31:0]
    g1[6] = 0;                         // stride hi | tensor_dim1_stride lo (unused)
    g1[7] = 0;
    i32x4 z4 = {0, 0, 0, 0};
    i32x8 z8 = {0, 0, 0, 0, 0, 0, 0, 0};
    __builtin_amdgcn_tensor_load_to_lds(g0, g1, z4, z4, z8, 0);
}

// ---------------------------------------------------------------------------
// Kernel 0: pack w_trans [512,256,3,3] f32 -> Wp[n][k] bf16, k=(ky*3+kx)*256+ci
// ---------------------------------------------------------------------------
__global__ __launch_bounds__(256) void pack_w_kernel(
        const float* __restrict__ w, unsigned short* __restrict__ Wp) {
    int i = blockIdx.x * 256 + threadIdx.x;
    if (i < 512 * 2304) {
        int n  = i / 2304;
        int k  = i % 2304;
        int r  = k >> 8;           // (ky*3+kx)
        int ci = k & 255;
        int ky = r / 3, kx = r % 3;
        float v = w[((n * 256 + ci) * 3 + ky) * 3 + kx];
        Wp[i] = f2bf(v);
    }
}

// ---------------------------------------------------------------------------
// Kernel 1: fused per level: 3x3 conv (WMMA bf16 implicit GEMM, TDM-staged
// weights, double-buffered) + bias + ReLU (LDS only) -> 1x1 cls/reg ->
// sigmoid/decode/clip -> d_out + scores/boxes workspace.
// One workgroup = 32 positions x all 512 channels; 8 waves x 64 N each.
// ---------------------------------------------------------------------------
__global__ __launch_bounds__(256) void conv_fused_kernel(
        const float* __restrict__ f,          // [B,256,H,W]
        const unsigned short* __restrict__ Wp,// [512][2304] bf16
        const float* __restrict__ bias_t,     // [512]
        const float* __restrict__ wcls,       // [3,512]
        const float* __restrict__ bcls,       // [3]
        const float* __restrict__ wreg,       // [12,512]
        const float* __restrict__ breg,       // [12]
        const float* __restrict__ img_info,   // [B,3]
        float* __restrict__ out_cls,          // d_out + CLS_BASE
        float* __restrict__ out_reg,          // d_out + REG_BASE
        float* __restrict__ scoresBuf,        // [2*TOTA]
        float* __restrict__ boxesBuf,         // [2*TOTA*4]
        int H, int Wshift, int stride, int sizeb, int offA) {

    const int W   = 1 << Wshift;
    const int tid = threadIdx.x;
    const int b   = blockIdx.y;
    const int mt  = blockIdx.x;

    // LDS arena: As[2] (2x2560B) | Bs[2] (2x40960B); Tl aliases the (dead
    // after the K-loop) Bs region.  Total 87040 B.
    __shared__ __align__(16) unsigned char smem[87040];
    typedef unsigned short AsTile[32][40];
    typedef unsigned short BsTile[512][40];
    AsTile* As = (AsTile*)smem;                                    // As[0..1]
    BsTile* Bs = (BsTile*)(smem + 5120);                           // Bs[0..1]
    unsigned short (*Tl)[520] = (unsigned short (*)[520])(smem + 5120);

    const int wid   = tid >> 5;
    const int lane  = tid & 31;
    const int ln    = lane & 15;
    const int lhalf = (lane >= 16);

    v8f acc[2][4] = {};

    // ---- prologue: stage k-slice 0 (A via im2col stores, B via TDM) ----
    if (wid == 0)
        tdm_load_B(Wp, (unsigned)(size_t)(smem + 5120));
    {
        #pragma unroll
        for (int q = 0; q < 4; ++q) {
            int idx  = tid + q * 256;
            int mloc = idx & 31;
            int dk   = idx >> 5;          // ci (kbase=0 -> ky=kx=0)
            int mg   = mt * 32 + mloc;
            int y    = mg >> Wshift;
            int x    = mg & (W - 1);
            int iy   = y - 1, ix = x - 1;
            float v = 0.f;
            if (iy >= 0 && ix >= 0)
                v = f[((((size_t)b * 256 + dk) * H + iy) << Wshift) + ix];
            As[0][mloc][dk] = f2bf(v);
        }
    }

    for (int kk = 0; kk < 72; ++kk) {
        const int cur = kk & 1;
        // wave 0 tracks the outstanding TDM for Bs[cur]; everyone else
        // synchronizes on the workgroup barrier.
        if (wid == 0)
            __builtin_amdgcn_s_wait_tensorcnt(0);
        __syncthreads();          // Bs[cur] + As[cur] visible to all waves

        // ---- issue next slice into the other buffer (overlaps compute) ----
        if (kk + 1 < 72) {
            const int nxt = cur ^ 1;
            if (wid == 0)
                tdm_load_B(Wp + (size_t)(kk + 1) * 32,
                           (unsigned)(size_t)(smem + 5120 + nxt * 40960));
            const int kb2 = (kk + 1) * 32;
            const int r2  = kb2 >> 8;
            const int ci2 = kb2 & 255;
            const int ky2 = r2 / 3, kx2 = r2 % 3;
            #pragma unroll
            for (int q = 0; q < 4; ++q) {
                int idx  = tid + q * 256;
                int mloc = idx & 31;
                int dk   = idx >> 5;
                int mg   = mt * 32 + mloc;
                int y    = mg >> Wshift;
                int x    = mg & (W - 1);
                int iy   = y + ky2 - 1;
                int ix   = x + kx2 - 1;
                float v = 0.f;
                if (iy >= 0 && iy < H && ix >= 0 && ix < W)
                    v = f[((((size_t)b * 256 + ci2 + dk) * H + iy) << Wshift) + ix];
                As[nxt][mloc][dk] = f2bf(v);
            }
        }

        // ---- A fragments (ISA 16-bit A 16x32 lane layout) ----
        Frag afr[2];
        #pragma unroll
        for (int mi = 0; mi < 2; ++mi) {
            int m = mi * 16 + ln;
            #pragma unroll
            for (int j = 0; j < 8; ++j) {
                int K = ((j < 4) ? 2 * j : 2 * j + 8) + (lhalf ? 8 : 0);
                afr[mi].d[j] = *(const unsigned int*)&As[cur][m][K];
            }
        }
        // ---- B fragments + 8 WMMAs ----
        #pragma unroll
        for (int ni = 0; ni < 4; ++ni) {
            int n = wid * 64 + ni * 16 + ln;
            Frag bfr;
            #pragma unroll
            for (int j = 0; j < 8; ++j)
                bfr.d[j] = *(const unsigned int*)&Bs[cur][n][((j < 4) ? 2 * j : 2 * j + 8) + (lhalf ? 8 : 0)];
            #pragma unroll
            for (int mi = 0; mi < 2; ++mi)
                acc[mi][ni] = wmma_bf16(afr[mi].b, bfr.b, acc[mi][ni]);
        }
        __syncthreads();          // readers done before buffer is re-staged
    }

    // ---- bias + ReLU, stash t tile in LDS (aliases dead Bs region) ----
    #pragma unroll
    for (int mi = 0; mi < 2; ++mi) {
        #pragma unroll
        for (int ni = 0; ni < 4; ++ni) {
            int n = wid * 64 + ni * 16 + ln;
            float bv = bias_t[n];
            #pragma unroll
            for (int j = 0; j < 8; ++j) {
                int m = mi * 16 + j + (lhalf ? 8 : 0);
                float v = acc[mi][ni][j] + bv;
                Tl[m][n] = f2bf(fmaxf(v, 0.f));
            }
        }
    }
    __syncthreads();

    // ---- fused 1x1 cls/reg + sigmoid + anchor decode + clip ----
    if (tid < 96) {
        int mloc = tid / 3;
        int a    = tid % 3;
        int mg   = mt * 32 + mloc;

        float dc = bcls[a];
        float d0 = breg[4 * a + 0], d1 = breg[4 * a + 1];
        float d2 = breg[4 * a + 2], d3 = breg[4 * a + 3];
        const float* wc  = wcls + a * 512;
        const float* wr0 = wreg + (4 * a + 0) * 512;
        const float* wr1 = wreg + (4 * a + 1) * 512;
        const float* wr2 = wreg + (4 * a + 2) * 512;
        const float* wr3 = wreg + (4 * a + 3) * 512;
        for (int k = 0; k < 512; ++k) {
            float t = bf2f(Tl[mloc][k]);
            dc += t * wc[k];
            d0 += t * wr0[k]; d1 += t * wr1[k];
            d2 += t * wr2[k]; d3 += t * wr3[k];
        }

        int aidx = b * TOTA + offA + mg * 3 + a;
        out_cls[aidx] = dc;
        float* orp = out_reg + (size_t)aidx * 4;
        orp[0] = d0; orp[1] = d1; orp[2] = d2; orp[3] = d3;

        float score = 1.f / (1.f + __expf(-dc));
        scoresBuf[aidx] = score;

        // anchors (py-faster-rcnn style, SCALES={8})
        float ratio = (a == 0) ? 0.5f : (a == 1 ? 1.f : 2.f);
        float fsb = (float)sizeb;
        float wsr = roundf(sqrtf(fsb * fsb / ratio));
        float hsr = roundf(wsr * ratio);
        float wa = wsr * 8.f, ha = hsr * 8.f;
        float ctr = (fsb - 1.f) * 0.5f;
        int y = mg >> Wshift, x = mg & (W - 1);
        float x1 = (float)(x * stride) + ctr - 0.5f * (wa - 1.f);
        float y1 = (float)(y * stride) + ctr - 0.5f * (ha - 1.f);
        float xa = x1 + 0.5f * wa;              // legacy (+1) widths
        float ya = y1 + 0.5f * ha;
        float cx = d0 * wa + xa, cy = d1 * ha + ya;
        float pw = __expf(d2) * wa, ph = __expf(d3) * ha;

        float ih = img_info[b * 3 + 0] - 1.f;
        float iw = img_info[b * 3 + 1] - 1.f;
        float bx1 = fminf(fmaxf(cx - 0.5f * pw, 0.f), iw);
        float by1 = fminf(fmaxf(cy - 0.5f * ph, 0.f), ih);
        float bx2 = fminf(fmaxf(cx + 0.5f * pw, 0.f), iw);
        float by2 = fminf(fmaxf(cy + 0.5f * ph, 0.f), ih);
        float* bp = boxesBuf + (size_t)aidx * 4;
        bp[0] = bx1; bp[1] = by1; bp[2] = bx2; bp[3] = by2;
    }
}

// ---------------------------------------------------------------------------
// Kernel 2: per-(level,batch) NMS.  histogram-threshold top-k selection,
// in-LDS bitonic sort (1024), greedy suppression, write top-300 survivors.
// ---------------------------------------------------------------------------
__global__ __launch_bounds__(1024) void nms_kernel(
        const float* __restrict__ scoresBuf, const float* __restrict__ boxesBuf,
        float* __restrict__ lvlScores,       // [B][1500]
        float* __restrict__ lvlBoxes) {      // [B][1500][4]
    const int HWs[5]   = {16384, 4096, 1024, 256, 64};
    const int offAs[5] = {0, 49152, 61440, 64512, 65280};
    const int l = blockIdx.x, b = blockIdx.y;
    const int N = HWs[l] * 3;
    const int base = b * TOTA + offAs[l];
    const int k = N < 1000 ? N : 1000;
    const int tid = threadIdx.x;

    __shared__ float sc[1024];
    __shared__ int   id[1024];
    __shared__ float bx[1024][4];
    __shared__ unsigned int hist[1024];
    __shared__ int keepArr[1024];
    __shared__ int pos[1024];
    __shared__ int selA, selT, s_tb, s_above;

    if (tid < 300) {                       // pre-fill output slots (padding)
        int o = b * 1500 + l * 300 + tid;
        lvlScores[o] = -INF_F;
        lvlBoxes[o * 4 + 0] = 0.f; lvlBoxes[o * 4 + 1] = 0.f;
        lvlBoxes[o * 4 + 2] = 0.f; lvlBoxes[o * 4 + 3] = 0.f;
    }
    hist[tid] = 0; sc[tid] = -INF_F; id[tid] = -1;
    __syncthreads();

    for (int i = tid; i < N; i += 1024) {
        float s = scoresBuf[base + i];
        int bk = (int)(s * 1024.f);
        bk = bk < 0 ? 0 : (bk > 1023 ? 1023 : bk);
        atomicAdd(&hist[bk], 1u);
    }
    __syncthreads();
    if (tid == 0) {
        unsigned c = 0; int t = 0;
        for (int bk = 1023; bk >= 0; --bk) {
            unsigned nc = c + hist[bk];
            t = bk;
            if (nc >= (unsigned)k) break;
            c = nc;
        }
        s_tb = t; s_above = (int)c; selA = 0; selT = 0;
    }
    __syncthreads();
    const int TB = s_tb, AC = s_above;

    for (int i = tid; i < N; i += 1024) {
        float s = scoresBuf[base + i];
        int bk = (int)(s * 1024.f);
        bk = bk < 0 ? 0 : (bk > 1023 ? 1023 : bk);
        int slot = -1;
        if (bk > TB)      slot = atomicAdd(&selA, 1);
        else if (bk == TB) {
            int t2 = AC + atomicAdd(&selT, 1);
            if (t2 < 1024) slot = t2;
        }
        if (slot >= 0 && slot < 1024) { sc[slot] = s; id[slot] = i; }
    }
    __syncthreads();

    // bitonic sort 1024, descending by score
    for (int ksz = 2; ksz <= 1024; ksz <<= 1)
        for (int j = ksz >> 1; j > 0; j >>= 1) {
            int ix = tid ^ j;
            if (ix > tid) {
                bool up = (tid & ksz) == 0;
                float a = sc[tid], c2 = sc[ix];
                bool sw = up ? (a < c2) : (a > c2);
                if (sw) {
                    sc[tid] = c2; sc[ix] = a;
                    int ta = id[tid]; id[tid] = id[ix]; id[ix] = ta;
                }
            }
            __syncthreads();
        }

    bool valid = (tid < k) && (id[tid] >= 0) && (sc[tid] > -1e30f);
    float mx1 = 0.f, my1 = 0.f, mx2 = 0.f, my2 = 0.f;
    if (valid) {
        const float* bp = boxesBuf + (size_t)(base + id[tid]) * 4;
        mx1 = bp[0]; my1 = bp[1]; mx2 = bp[2]; my2 = bp[3];
    }
    bx[tid][0] = mx1; bx[tid][1] = my1; bx[tid][2] = mx2; bx[tid][3] = my2;
    keepArr[tid] = valid ? 1 : 0;
    __syncthreads();

    float areaJ = (mx2 - mx1 + 1.f) * (my2 - my1 + 1.f);
    for (int i = 0; i < k - 1; ++i) {
        if (keepArr[i] && tid > i && keepArr[tid]) {
            float ix1 = fmaxf(bx[i][0], mx1);
            float iy1 = fmaxf(bx[i][1], my1);
            float ix2 = fminf(bx[i][2], mx2);
            float iy2 = fminf(bx[i][3], my2);
            float iww = fmaxf(ix2 - ix1 + 1.f, 0.f);
            float ihh = fmaxf(iy2 - iy1 + 1.f, 0.f);
            float inter = iww * ihh;
            float ai = (bx[i][2] - bx[i][0] + 1.f) * (bx[i][3] - bx[i][1] + 1.f);
            float iou = inter / (ai + areaJ - inter);
            if (iou > 0.7f) keepArr[tid] = 0;
        }
        __syncthreads();
    }

    // prefix-sum over keep flags -> compact top-300 survivors in sorted order
    pos[tid] = keepArr[tid];
    __syncthreads();
    for (int off = 1; off < 1024; off <<= 1) {
        int v = (tid >= off) ? pos[tid - off] : 0;
        __syncthreads();
        pos[tid] += v;
        __syncthreads();
    }
    if (keepArr[tid]) {
        int sl = pos[tid] - 1;
        if (sl < 300) {
            int o = b * 1500 + l * 300 + sl;
            lvlScores[o] = sc[tid];
            lvlBoxes[o * 4 + 0] = mx1; lvlBoxes[o * 4 + 1] = my1;
            lvlBoxes[o * 4 + 2] = mx2; lvlBoxes[o * 4 + 3] = my2;
        }
    }
}

// ---------------------------------------------------------------------------
// Kernel 3: per-batch final top-300 over 5*300 candidates -> rois [B,300,5]
// ---------------------------------------------------------------------------
__global__ __launch_bounds__(1024) void topk_final_kernel(
        const float* __restrict__ lvlScores, const float* __restrict__ lvlBoxes,
        float* __restrict__ rois) {
    const int b = blockIdx.x, tid = threadIdx.x;
    __shared__ float s2[2048];
    __shared__ int   i2[2048];
    for (int i = tid; i < 2048; i += 1024) {
        if (i < 1500) { s2[i] = lvlScores[b * 1500 + i]; i2[i] = i; }
        else          { s2[i] = -INF_F; i2[i] = -1; }
    }
    __syncthreads();
    for (int ksz = 2; ksz <= 2048; ksz <<= 1)
        for (int j = ksz >> 1; j > 0; j >>= 1) {
            for (int i = tid; i < 2048; i += 1024) {
                int ix = i ^ j;
                if (ix > i) {
                    bool up = (i & ksz) == 0;
                    float a = s2[i], c2 = s2[ix];
                    bool sw = up ? (a < c2) : (a > c2);
                    if (sw) {
                        s2[i] = c2; s2[ix] = a;
                        int t = i2[i]; i2[i] = i2[ix]; i2[ix] = t;
                    }
                }
            }
            __syncthreads();
        }
    if (tid < 300) {
        float s = s2[tid];
        float* rp = rois + (size_t)(b * 300 + tid) * 5;
        rp[0] = (float)b;
        if (s > -1e30f && i2[tid] >= 0) {
            const float* bp = lvlBoxes + (size_t)(b * 1500 + i2[tid]) * 4;
            rp[1] = bp[0]; rp[2] = bp[1]; rp[3] = bp[2]; rp[4] = bp[3];
        } else {
            rp[1] = 0.f; rp[2] = 0.f; rp[3] = 0.f; rp[4] = 0.f;
        }
    }
}

// ---------------------------------------------------------------------------
extern "C" void kernel_launch(void* const* d_in, const int* in_sizes, int n_in,
                              void* d_out, int out_size, void* d_ws, size_t ws_size,
                              hipStream_t stream) {
    (void)in_sizes; (void)n_in; (void)out_size; (void)ws_size;
    const float* f[5] = {(const float*)d_in[0], (const float*)d_in[1],
                         (const float*)d_in[2], (const float*)d_in[3],
                         (const float*)d_in[4]};
    const float* img_info = (const float*)d_in[6];
    const float* w_trans  = (const float*)d_in[8];
    const float* b_trans  = (const float*)d_in[9];
    const float* w_cls    = (const float*)d_in[10];
    const float* b_cls    = (const float*)d_in[11];
    const float* w_reg    = (const float*)d_in[12];
    const float* b_reg    = (const float*)d_in[13];
    float* out = (float*)d_out;

    char* ws = (char*)d_ws;
    unsigned short* Wp     = (unsigned short*)ws;                    // 2,359,296 B
    float* scoresBuf = (float*)(ws + 2359296);                       //   523,776 B
    float* boxesBuf  = (float*)(ws + 2359296 + 523776);              // 2,095,104 B
    float* lvlScores = (float*)(ws + 2359296 + 523776 + 2095104);    //    12,000 B
    float* lvlBoxes  = lvlScores + 3000;                             //    48,000 B

    pack_w_kernel<<<dim3(4608), dim3(256), 0, stream>>>(w_trans, Wp);

    const int Hs[5]      = {128, 64, 32, 16, 8};
    const int wsh[5]     = {7, 6, 5, 4, 3};
    const int strides[5] = {4, 8, 16, 32, 64};
    const int sbs[5]     = {4, 8, 16, 32, 64};
    const int offAs[5]   = {0, 49152, 61440, 64512, 65280};
    for (int l = 0; l < 5; ++l) {
        int HW = Hs[l] * Hs[l];
        conv_fused_kernel<<<dim3(HW / 32, 2), dim3(256), 0, stream>>>(
            f[l], Wp, b_trans, w_cls, b_cls, w_reg, b_reg, img_info,
            out + CLS_BASE, out + REG_BASE, scoresBuf, boxesBuf,
            Hs[l], wsh[l], strides[l], sbs[l], offAs[l]);
    }
    nms_kernel<<<dim3(5, 2), dim3(1024), 0, stream>>>(scoresBuf, boxesBuf,
                                                      lvlScores, lvlBoxes);
    topk_final_kernel<<<dim3(2), dim3(1024), 0, stream>>>(lvlScores, lvlBoxes, out);
}
